// LightMoE_47038481825970
// MI455X (gfx1250) — matrix operand order repaired and checked by reading, not compiled, not run on hardware
//
#include <hip/hip_runtime.h>
#include <math.h>

typedef __attribute__((ext_vector_type(16))) __bf16 v16bf;
typedef __attribute__((ext_vector_type(8)))  __bf16 v8bf;
typedef __attribute__((ext_vector_type(8)))  float  v8f;

#define B_   64
#define F_   128
#define L_   512
#define P_   336
#define FF_  2048
#define E_   8
#define TOPK 4

#define MT    32          // rows per workgroup
#define CH    256         // FF chunk held in LDS
#define NCH   (FF_/CH)    // 8 chunks
#define XPAD  8
#define HPAD  8

// workspace layout (bytes)
#define WS_GATES_OFF  0
#define WS_GATES_SZ   ((size_t)F_ * B_ * E_ * 4)               // 262144
#define WS_FCW_OFF    (WS_GATES_OFF + WS_GATES_SZ)
#define WS_FCW_SZ     ((size_t)L_ * FF_ * 2)                   // 2 MB
#define WS_EXPW_OFF   (WS_FCW_OFF + WS_FCW_SZ)
#define WS_EXPW_SZ    ((size_t)E_ * FF_ * P_ * 2)              // 10.5 MB
#define WS_NEED       (WS_EXPW_OFF + WS_EXPW_SZ)

// ---------------------------------------------------------------------------
// Branch-free exact GELU: 0.5x(1+erf(x/sqrt2)), erf via A&S 7.1.26 (|e|<1.5e-7)
// ---------------------------------------------------------------------------
__device__ __forceinline__ float gelu_exact(float v) {
    const float u = 0.70710678118654752f * v;
    const float y = __builtin_fabsf(u);
    const float t = __builtin_amdgcn_rcpf(__builtin_fmaf(0.3275911f, y, 1.0f));
    float p = __builtin_fmaf(t, 1.061405429f, -1.453152027f);
    p = __builtin_fmaf(t, p, 1.421413741f);
    p = __builtin_fmaf(t, p, -0.284496736f);
    p = __builtin_fmaf(t, p, 0.254829592f);
    p *= t;
    const float ex = __builtin_amdgcn_exp2f(-y * y * 1.4426950408889634f);
    float er = __builtin_fmaf(-p, ex, 1.0f);
    er = __builtin_copysignf(er, u);
    return 0.5f * v * (1.0f + er);
}

// ---------------------------------------------------------------------------
// Kernel 0: zero the loss accumulator slot
// ---------------------------------------------------------------------------
__global__ void zero_loss_kernel(float* loss) {
    if (threadIdx.x == 0) *loss = 0.0f;
}

// ---------------------------------------------------------------------------
// Pre-pass: f32 -> bf16 weight conversion (8 elements / thread)
// ---------------------------------------------------------------------------
__global__ __launch_bounds__(256) void cvt_bf16_kernel(
    const float* __restrict__ src, __bf16* __restrict__ dst, int n8)
{
    const int i = blockIdx.x * 256 + threadIdx.x;
    if (i < n8) {
        float4 v0 = ((const float4*)src)[i * 2 + 0];
        float4 v1 = ((const float4*)src)[i * 2 + 1];
        v8bf o;
        o[0] = (__bf16)v0.x; o[1] = (__bf16)v0.y; o[2] = (__bf16)v0.z; o[3] = (__bf16)v0.w;
        o[4] = (__bf16)v1.x; o[5] = (__bf16)v1.y; o[6] = (__bf16)v1.z; o[7] = (__bf16)v1.w;
        ((v8bf*)dst)[i] = o;
    }
}

// ---------------------------------------------------------------------------
// Kernel 1: gating (logits -> TopK adaptive gating) + load-balance loss.
// ---------------------------------------------------------------------------
__global__ __launch_bounds__(64) void gate_kernel(
    const float* __restrict__ te,      // [B,F,L]
    const float* __restrict__ gate_w,  // [L,E]
    const float* __restrict__ gate_b,  // [E]
    float* __restrict__ gates_out,     // [F,B,E] (workspace)
    float* __restrict__ loss_out)      // scalar
{
    __shared__ float gw[L_ * E_];
    __shared__ float red[64];
    __shared__ float glds[64 * E_];

    const int f = blockIdx.x;
    const int b = threadIdx.x;

    for (int i = threadIdx.x; i < L_ * E_ / 4; i += 64)
        ((float4*)gw)[i] = ((const float4*)gate_w)[i];
    __syncthreads();

    float lg[E_];
    #pragma unroll
    for (int e = 0; e < E_; ++e) lg[e] = gate_b[e];

    const float* tr = te + ((size_t)b * F_ + f) * L_;
    for (int l4 = 0; l4 < L_ / 4; ++l4) {
        float4 t = ((const float4*)tr)[l4];
        int l = l4 * 4;
        #pragma unroll
        for (int e = 0; e < E_; ++e) {
            lg[e] += t.x * gw[(l + 0) * E_ + e] + t.y * gw[(l + 1) * E_ + e]
                   + t.z * gw[(l + 2) * E_ + e] + t.w * gw[(l + 3) * E_ + e];
        }
    }

    float mx = lg[0];
    #pragma unroll
    for (int e = 1; e < E_; ++e) mx = fmaxf(mx, lg[e]);
    float sm[E_]; float s = 0.f;
    #pragma unroll
    for (int e = 0; e < E_; ++e) { sm[e] = expf(lg[e] - mx); s += sm[e]; }
    #pragma unroll
    for (int e = 0; e < E_; ++e) sm[e] /= s;

    // kth value: sorted ascending index (E - TOPK)
    float srt[E_];
    #pragma unroll
    for (int e = 0; e < E_; ++e) srt[e] = lg[e];
    for (int i = 1; i < E_; ++i) {
        float v = srt[i]; int j = i - 1;
        while (j >= 0 && srt[j] > v) { srt[j + 1] = srt[j]; --j; }
        srt[j + 1] = v;
    }
    const float kth = srt[E_ - TOPK];

    float ss = 0.f;
    #pragma unroll
    for (int e = 0; e < E_; ++e) { float d = sm[e] - 0.125f; ss += d * d; }
    const float stdv = sqrtf(ss / (float)(E_ - 1));

    red[b] = stdv;
    __syncthreads();
    float sstd = 0.f;
    for (int i = 0; i < 64; ++i) sstd += red[i];
    const float alpha = (sstd / 64.0f) / (stdv + 1e-6f);   // BASE_ALPHA = 1

    float dec[E_]; float dmx = -1e30f;
    #pragma unroll
    for (int e = 0; e < E_; ++e) {
        dec[e] = (lg[e] < kth) ? alpha * log1pf(sm[e]) : alpha * expm1f(sm[e]);
        dmx = fmaxf(dmx, dec[e]);
    }
    float ds = 0.f; float g[E_];
    #pragma unroll
    for (int e = 0; e < E_; ++e) { g[e] = expf(dec[e] - dmx); ds += g[e]; }
    #pragma unroll
    for (int e = 0; e < E_; ++e) {
        g[e] /= ds;
        glds[b * E_ + e] = g[e];
        gates_out[((size_t)f * B_ + b) * E_ + e] = g[e];
    }
    __syncthreads();

    if (threadIdx.x == 0) {
        float imp[E_]; float msum = 0.f;
        #pragma unroll
        for (int e = 0; e < E_; ++e) {
            float a = 0.f;
            for (int bb = 0; bb < 64; ++bb) a += glds[bb * E_ + e];
            imp[e] = a; msum += a;
        }
        const float mn = msum / (float)E_;
        float vs = 0.f;
        #pragma unroll
        for (int e = 0; e < E_; ++e) { float d = imp[e] - mn; vs += d * d; }
        const float var = vs * (float)P_ / (float)(E_ * P_ - 1);
        atomicAdd(loss_out, var / (mn * mn + 1e-10f));
    }
}

// ---------------------------------------------------------------------------
// B-fragment loader: 32x16 bf16 fragment, lane holds col n=lane%16, k = 8h+s (s<8) / 16+8h+(s-8).
// ---------------------------------------------------------------------------
template <bool PRE>
__device__ __forceinline__ v16bf load_bfrag_f(const float* bp32, const __bf16* bp16, size_t ld) {
    v16bf bf;
    #pragma unroll
    for (int s = 0; s < 16; ++s) {
        const size_t o = (size_t)(s < 8 ? s : s + 8) * ld;
        if constexpr (PRE) bf[s] = bp16[o];
        else               bf[s] = (__bf16)bp32[o];
    }
    return bf;
}

// ---------------------------------------------------------------------------
// Kernel 2: fused  h = gelu(x@fc_w+b)  ->  out = sum_e g_e*(h@W_e + b_e)
// ---------------------------------------------------------------------------
template <bool PRE>
__global__ __launch_bounds__(256) void moe_kernel(
    const float* __restrict__ x,        // [B,F,L]
    const float* __restrict__ fc_w,     // [L,FF]  (f32, fallback)
    const __bf16* __restrict__ fcw_bf,  // [L,FF]  (bf16, PRE)
    const float* __restrict__ fc_b,     // [FF]
    const float* __restrict__ exp_w,    // [E,FF,P] (f32, fallback)
    const __bf16* __restrict__ expw_bf, // [E,FF,P] (bf16, PRE)
    const float* __restrict__ exp_b,    // [E,P]
    const float* __restrict__ gates,    // [F,B,E]
    float* __restrict__ out)            // [B,F,P]
{
    __shared__ __bf16 xl[MT][L_ + XPAD];   // 33280 B
    __shared__ __bf16 hl[MT][CH + HPAD];   // 16896 B
    __shared__ float  gl[MT][E_];          // 1024 B

    const int tid   = threadIdx.x;
    const int wave  = tid >> 5;
    const int lane  = tid & 31;
    const int m16   = lane & 15;
    const int khalf = lane >> 4;          // K-half select per ISA 16-bit A layout
    const int mBase = blockIdx.x * MT;

    // ---- load x M-tile as bf16 into LDS ----
    {
        const int r    = tid >> 3;
        const int cgrp = tid & 7;
        const int row  = mBase + r;
        const int f = row >> 6, b = row & 63;
        const float4* src = (const float4*)(x + ((size_t)b * F_ + f) * L_);
        #pragma unroll
        for (int i = 0; i < 16; ++i) {
            const int c4 = cgrp + i * 8;
            float4 v = src[c4];
            const int c = c4 * 4;
            xl[r][c + 0] = (__bf16)v.x; xl[r][c + 1] = (__bf16)v.y;
            xl[r][c + 2] = (__bf16)v.z; xl[r][c + 3] = (__bf16)v.w;
        }
    }
    if (tid < MT * E_) {
        const int r = tid >> 3, e = tid & 7;
        gl[r][e] = gates[(size_t)(mBase + r) * E_ + e];
    }
    __syncthreads();

    // persistent output accumulators: up to 3 P-tiles per wave
    float oacc[3][2][8];
    #pragma unroll
    for (int t = 0; t < 3; ++t)
        for (int sI = 0; sI < 2; ++sI)
            for (int r = 0; r < 8; ++r) oacc[t][sI][r] = 0.f;

    for (int chunk = 0; chunk < NCH; ++chunk) {
        const int nBaseC = chunk * CH;

        // ===== phase 1: h chunk = gelu(x @ fc_w[:, chunk] + fc_b) =====
        // each wave computes both of its N-tiles per K pass (A reuse: 4 wmma / A-load)
        {
            const int n0a = nBaseC + wave * 32;
            const int n0b = n0a + 16;
            v8f c0a = {}, c1a = {}, c0b = {}, c1b = {};
            #pragma unroll 2
            for (int ks = 0; ks < L_ / 32; ++ks) {
                const int k0 = ks * 32;
                v8bf lo0 = *(const v8bf*)&xl[m16][k0 + khalf * 8];
                v8bf hi0 = *(const v8bf*)&xl[m16][k0 + 16 + khalf * 8];
                v8bf lo1 = *(const v8bf*)&xl[m16 + 16][k0 + khalf * 8];
                v8bf hi1 = *(const v8bf*)&xl[m16 + 16][k0 + 16 + khalf * 8];
                v16bf a0 = __builtin_shufflevector(lo0, hi0, 0,1,2,3,4,5,6,7,8,9,10,11,12,13,14,15);
                v16bf a1 = __builtin_shufflevector(lo1, hi1, 0,1,2,3,4,5,6,7,8,9,10,11,12,13,14,15);
                const size_t rowoff = (size_t)(k0 + khalf * 8) * FF_ + m16;
                v16bf ba = load_bfrag_f<PRE>(fc_w + rowoff + n0a, fcw_bf + rowoff + n0a, FF_);
                v16bf bb = load_bfrag_f<PRE>(fc_w + rowoff + n0b, fcw_bf + rowoff + n0b, FF_);
                c0a = __builtin_amdgcn_wmma_f32_16x16x32_bf16(false, a0, false, ba, (short)0, c0a, false, false);
                c1a = __builtin_amdgcn_wmma_f32_16x16x32_bf16(false, a1, false, ba, (short)0, c1a, false, false);
                c0b = __builtin_amdgcn_wmma_f32_16x16x32_bf16(false, a0, false, bb, (short)0, c0b, false, false);
                c1b = __builtin_amdgcn_wmma_f32_16x16x32_bf16(false, a1, false, bb, (short)0, c1b, false, false);
            }
            const int n = lane & 15;
            const float biasa = fc_b[n0a + n];
            const float biasb = fc_b[n0b + n];
            #pragma unroll
            for (int r = 0; r < 8; ++r) {
                const int ml = r + 8 * khalf;
                hl[ml][(n0a - nBaseC) + n]      = (__bf16)gelu_exact(c0a[r] + biasa);
                hl[ml + 16][(n0a - nBaseC) + n] = (__bf16)gelu_exact(c1a[r] + biasa);
                hl[ml][(n0b - nBaseC) + n]      = (__bf16)gelu_exact(c0b[r] + biasb);
                hl[ml + 16][(n0b - nBaseC) + n] = (__bf16)gelu_exact(c1b[r] + biasb);
            }
        }
        __syncthreads();

        // ===== phase 2: experts (paired, 2 per K pass) on this K chunk =====
        int nti = 0;
        for (int ntile = wave; ntile < P_ / 16; ntile += 8, ++nti) {
            const int n0 = ntile * 16;
            #pragma unroll 1
            for (int ep = 0; ep < E_ / 2; ++ep) {
                const int e0 = ep * 2, e1 = e0 + 1;
                const size_t wo0 = ((size_t)e0 * FF_ + nBaseC) * P_ + n0;
                const size_t wo1 = wo0 + (size_t)FF_ * P_;
                v8f c00 = {}, c10 = {}, c01 = {}, c11 = {};
                #pragma unroll 2
                for (int ks = 0; ks < CH / 32; ++ks) {
                    const int k0 = ks * 32;
                    v8bf lo0 = *(const v8bf*)&hl[m16][k0 + khalf * 8];
                    v8bf hi0 = *(const v8bf*)&hl[m16][k0 + 16 + khalf * 8];
                    v8bf lo1 = *(const v8bf*)&hl[m16 + 16][k0 + khalf * 8];
                    v8bf hi1 = *(const v8bf*)&hl[m16 + 16][k0 + 16 + khalf * 8];
                    v16bf a0 = __builtin_shufflevector(lo0, hi0, 0,1,2,3,4,5,6,7,8,9,10,11,12,13,14,15);
                    v16bf a1 = __builtin_shufflevector(lo1, hi1, 0,1,2,3,4,5,6,7,8,9,10,11,12,13,14,15);
                    const size_t rowoff = (size_t)(k0 + khalf * 8) * P_ + m16;
                    if constexpr (PRE)
                        __builtin_prefetch(expw_bf + wo0 + rowoff + (size_t)32 * P_, 0, 0);
                    else
                        __builtin_prefetch(exp_w + wo0 + rowoff + (size_t)32 * P_, 0, 0);
                    v16bf b0 = load_bfrag_f<PRE>(exp_w + wo0 + rowoff, expw_bf + wo0 + rowoff, P_);
                    v16bf b1 = load_bfrag_f<PRE>(exp_w + wo1 + rowoff, expw_bf + wo1 + rowoff, P_);
                    c00 = __builtin_amdgcn_wmma_f32_16x16x32_bf16(false, a0, false, b0, (short)0, c00, false, false);
                    c10 = __builtin_amdgcn_wmma_f32_16x16x32_bf16(false, a1, false, b0, (short)0, c10, false, false);
                    c01 = __builtin_amdgcn_wmma_f32_16x16x32_bf16(false, a0, false, b1, (short)0, c01, false, false);
                    c11 = __builtin_amdgcn_wmma_f32_16x16x32_bf16(false, a1, false, b1, (short)0, c11, false, false);
                }
                #pragma unroll
                for (int r = 0; r < 8; ++r) {
                    const int ml = r + 8 * khalf;
                    oacc[nti][0][r] += gl[ml][e0]      * c00[r] + gl[ml][e1]      * c01[r];
                    oacc[nti][1][r] += gl[ml + 16][e0] * c10[r] + gl[ml + 16][e1] * c11[r];
                }
            }
        }
        __syncthreads();
    }

    // ---- store: gate-weighted expert bias + transpose rows back to [B,F,P] ----
    int nti = 0;
    for (int ntile = wave; ntile < P_ / 16; ntile += 8, ++nti) {
        const int n0 = ntile * 16;
        const int p  = n0 + (lane & 15);
        #pragma unroll
        for (int r = 0; r < 8; ++r) {
            const int ml = r + 8 * khalf;
            #pragma unroll
            for (int sI = 0; sI < 2; ++sI) {
                const int m = ml + sI * 16;
                float gb = 0.f;
                #pragma unroll
                for (int e = 0; e < E_; ++e) gb += gl[m][e] * exp_b[e * P_ + p];
                const int row = mBase + m;
                const int f = row >> 6, b = row & 63;
                out[((size_t)b * F_ + f) * P_ + p] = oacc[nti][sI][r] + gb;
            }
        }
    }
}

// ---------------------------------------------------------------------------
extern "C" void kernel_launch(void* const* d_in, const int* in_sizes, int n_in,
                              void* d_out, int out_size, void* d_ws, size_t ws_size,
                              hipStream_t stream) {
    const float* x      = (const float*)d_in[0];
    const float* te     = (const float*)d_in[1];
    const float* gate_w = (const float*)d_in[2];
    const float* gate_b = (const float*)d_in[3];
    const float* fc_w   = (const float*)d_in[4];
    const float* fc_b   = (const float*)d_in[5];
    const float* exp_w  = (const float*)d_in[6];
    const float* exp_b  = (const float*)d_in[7];

    float* out  = (float*)d_out;                  // [B,F,P] then loss scalar
    float* loss = out + (size_t)B_ * F_ * P_;
    float* gates_ws = (float*)((char*)d_ws + WS_GATES_OFF);

    zero_loss_kernel<<<1, 32, 0, stream>>>(loss);
    gate_kernel<<<F_, 64, 0, stream>>>(te, gate_w, gate_b, gates_ws, loss);

    const int nblk = (F_ * B_) / MT;
    if (ws_size >= WS_NEED) {
        __bf16* fcw_bf  = (__bf16*)((char*)d_ws + WS_FCW_OFF);
        __bf16* expw_bf = (__bf16*)((char*)d_ws + WS_EXPW_OFF);
        const int n8_fc  = (L_ * FF_) / 8;
        const int n8_exp = (E_ * FF_ * P_) / 8;
        cvt_bf16_kernel<<<(n8_fc  + 255) / 256, 256, 0, stream>>>(fc_w,  fcw_bf,  n8_fc);
        cvt_bf16_kernel<<<(n8_exp + 255) / 256, 256, 0, stream>>>(exp_w, expw_bf, n8_exp);
        moe_kernel<true><<<nblk, 256, 0, stream>>>(x, fc_w, fcw_bf, fc_b,
                                                   exp_w, expw_bf, exp_b, gates_ws, out);
    } else {
        moe_kernel<false><<<nblk, 256, 0, stream>>>(x, fc_w, nullptr, fc_b,
                                                    exp_w, nullptr, exp_b, gates_ws, out);
    }
}
